// FlaxDbrxNormAttentionNorm_35553739276648
// MI455X (gfx1250) — matrix operand order repaired and checked by reading, not compiled, not run on hardware
//
#include <hip/hip_runtime.h>
#include <hip/hip_bf16.h>

// ---------------------------------------------------------------------------
// DBRX NormAttentionNorm for gfx1250 (MI455X), wave32 + WMMA bf16 +
// async-to-LDS double-buffered GEMM and attention (ASYNCcnt path).
// B=2, S=2048, D=2048, Hq=16, Hkv=4, hd=128, qkv_out=3072, clip=8, eps=1e-5
// ---------------------------------------------------------------------------

#define DB_B     2
#define DB_S     2048
#define DB_D     2048
#define DB_HQ    16
#define DB_HKV   4
#define DB_HD    128
#define DB_QKVN  3072          // D + 2*Hkv*hd
#define DB_M     (DB_B * DB_S) // 4096 rows
#define DB_EPS   1e-5f
#define DB_CLIP  8.0f
#define DB_NEGINF (-3.0e38f)

typedef __bf16  bf16_t;
typedef __attribute__((ext_vector_type(16))) __bf16 v16bf;
typedef __attribute__((ext_vector_type(8)))  float  v8f;

union FragU { v16bf v; unsigned int u[8]; };

__device__ __forceinline__ v8f vzero8() {
  v8f z;
#pragma unroll
  for (int i = 0; i < 8; ++i) z[i] = 0.0f;
  return z;
}

// A-matrix fragment (16x32 bf16, row-major source, row folded into ptr).
// Lane g = lane>>4:  VGPR v<4 : K = 8g+2v ;  VGPR v>=4 : K = 16+8g+2(v-4).
__device__ __forceinline__ v16bf load_a_frag(const bf16_t* __restrict__ rowp, int lane) {
  FragU f;
  const int g8 = (lane >> 4) * 8;
#pragma unroll
  for (int v = 0; v < 4; ++v)
    f.u[v] = *(const unsigned int*)(rowp + g8 + 2 * v);
#pragma unroll
  for (int v = 0; v < 4; ++v)
    f.u[4 + v] = *(const unsigned int*)(rowp + 16 + g8 + 2 * v);
  return f.v;
}

// B-matrix fragment (32x16 bf16). Source is K-contiguous per column.
// Lane g = lane>>4:  VGPR i<4 : K = 8g+2i ;  VGPR i>=4 : K = 16+8g+2(i-4).
__device__ __forceinline__ v16bf load_b_frag(const bf16_t* __restrict__ colp, int lane) {
  FragU f;
  const int kh = (lane >> 4) * 8;
#pragma unroll
  for (int i = 0; i < 8; ++i)
    f.u[i] = *(const unsigned int*)(colp + (i >> 2) * 16 + kh + 2 * (i & 3));
  return f.v;
}

// Async copy of 16 bytes (per lane) from global to LDS; tracked by ASYNCcnt.
__device__ __forceinline__ void async_load_b128(unsigned ldsoff, const void* gaddr) {
  asm volatile("global_load_async_to_lds_b128 %0, %1, off"
               :: "v"(ldsoff), "v"((unsigned long long)(uintptr_t)gaddr)
               : "memory");
}

__device__ __forceinline__ float hred_max16(float x) {
  x = fmaxf(x, __shfl_xor(x, 1, 32));
  x = fmaxf(x, __shfl_xor(x, 2, 32));
  x = fmaxf(x, __shfl_xor(x, 4, 32));
  x = fmaxf(x, __shfl_xor(x, 8, 32));
  return x;
}
__device__ __forceinline__ float hred_sum16(float x) {
  x += __shfl_xor(x, 1, 32);
  x += __shfl_xor(x, 2, 32);
  x += __shfl_xor(x, 4, 32);
  x += __shfl_xor(x, 8, 32);
  return x;
}

// ---------------------------------------------------------------------------
// Kernel 1: convert fp32 [K][N] weight -> bf16 [N][K] (transposed for B-frags)
// ---------------------------------------------------------------------------
__global__ void k_convtr(const float* __restrict__ in, bf16_t* __restrict__ out,
                         int K, int N) {
  size_t idx = (size_t)blockIdx.x * blockDim.x + threadIdx.x;
  if (idx >= (size_t)K * N) return;
  int n = (int)(idx % N);
  int k = (int)(idx / N);
  out[(size_t)n * K + k] = (bf16_t)in[idx];
}

// ---------------------------------------------------------------------------
// Kernel 2: RMSNorm fp32 row -> bf16 out   (one block per row of 2048)
// ---------------------------------------------------------------------------
__global__ void k_rmsnorm_bf16(const float* __restrict__ x, const float* __restrict__ w,
                               bf16_t* __restrict__ out) {
  __shared__ float red[8];
  const int row = blockIdx.x;
  const float* xr = x + (size_t)row * DB_D;
  float ss = 0.0f;
  for (int c = threadIdx.x; c < DB_D; c += 256) { float v = xr[c]; ss += v * v; }
#pragma unroll
  for (int m = 1; m < 32; m <<= 1) ss += __shfl_xor(ss, m, 32);
  if ((threadIdx.x & 31) == 0) red[threadIdx.x >> 5] = ss;
  __syncthreads();
  if (threadIdx.x == 0) {
    float t = 0.0f;
#pragma unroll
    for (int i = 0; i < 8; ++i) t += red[i];
    red[0] = rsqrtf(t * (1.0f / DB_D) + DB_EPS);
  }
  __syncthreads();
  const float rs = red[0];
  bf16_t* orow = out + (size_t)row * DB_D;
  for (int c = threadIdx.x; c < DB_D; c += 256)
    orow[c] = (bf16_t)(xr[c] * rs * w[c]);
}

// Same but fp32 output (final post-norm)
__global__ void k_rmsnorm_f32(const float* __restrict__ x, const float* __restrict__ w,
                              float* __restrict__ out) {
  __shared__ float red[8];
  const int row = blockIdx.x;
  const float* xr = x + (size_t)row * DB_D;
  float ss = 0.0f;
  for (int c = threadIdx.x; c < DB_D; c += 256) { float v = xr[c]; ss += v * v; }
#pragma unroll
  for (int m = 1; m < 32; m <<= 1) ss += __shfl_xor(ss, m, 32);
  if ((threadIdx.x & 31) == 0) red[threadIdx.x >> 5] = ss;
  __syncthreads();
  if (threadIdx.x == 0) {
    float t = 0.0f;
#pragma unroll
    for (int i = 0; i < 8; ++i) t += red[i];
    red[0] = rsqrtf(t * (1.0f / DB_D) + DB_EPS);
  }
  __syncthreads();
  const float rs = red[0];
  float* orow = out + (size_t)row * DB_D;
  for (int c = threadIdx.x; c < DB_D; c += 256)
    orow[c] = xr[c] * rs * w[c];
}

// ---------------------------------------------------------------------------
// Kernel 3: WMMA bf16 GEMM, async-LDS double-buffered B panel.
//   C[M,N] = A[M,K] @ B[K,N]   (B given transposed [N][K])
// Block = 8 waves = 256x64 C tile (wave = 32x64, 2 M-frags x 4 N-frags).
// B panel (64 cols x 32 k, 4 KB) staged in LDS via
// global_load_async_to_lds_b128 (prefetch distance 2, ASYNCcnt), shared by
// all 8 waves. A fragments register-prefetched one K-step ahead.
// ---------------------------------------------------------------------------
__global__ void __launch_bounds__(256)
k_gemm_bf16(const bf16_t* __restrict__ A, const bf16_t* __restrict__ BT,
            float* __restrict__ C, const float* __restrict__ residual,
            int M, int N, int K, int nblk64, int doClip) {
  __shared__ bf16_t Bbuf[2][64 * 32];   // [col][k] layout, 4 KB per buffer

  const int tid  = threadIdx.x;
  const int wid  = tid >> 5;
  const int lane = tid & 31;
  const int mb   = blockIdx.x / nblk64;
  const int nb   = blockIdx.x % nblk64;
  const int tm   = mb * 256 + wid * 32;
  const int tn   = nb * 64;
  const int nloc = lane & 15;
  const int hi   = lane >> 4;

  // This thread's 16-byte piece of the B panel: col = tid/4, k8 = (tid%4)*8
  const int bcol = tid >> 2;
  const int bk8  = (tid & 3) * 8;
  const bf16_t* bsrc = BT + (size_t)(tn + bcol) * K + bk8;
  const unsigned ldsoff0 = (unsigned)(uintptr_t)&Bbuf[0][bcol * 32 + bk8];
  const unsigned ldsoff1 = (unsigned)(uintptr_t)&Bbuf[1][bcol * 32 + bk8];

  v8f acc[2][4];
#pragma unroll
  for (int m = 0; m < 2; ++m)
#pragma unroll
    for (int j = 0; j < 4; ++j) acc[m][j] = vzero8();

  const bf16_t* Arow0 = A + (size_t)(tm + nloc) * K;
  const bf16_t* Arow1 = A + (size_t)(tm + 16 + nloc) * K;

  // Prime the pipeline: buffers 0 and 1 (K >= 64 always here)
  async_load_b128(ldsoff0, bsrc);
  async_load_b128(ldsoff1, bsrc + 32);
  v16bf a0 = load_a_frag(Arow0, lane);
  v16bf a1 = load_a_frag(Arow1, lane);

  for (int kb = 0; kb < K; kb += 32) {
    const int cur = (kb >> 5) & 1;
    // Wait for the oldest outstanding async fill (this iteration's buffer).
    if (kb + 32 < K) asm volatile("s_wait_asynccnt 0x1" ::: "memory");
    else             asm volatile("s_wait_asynccnt 0x0" ::: "memory");
    __syncthreads();                       // buf[cur] visible to all waves

    v16bf a0u = a0, a1u = a1;
    if (kb + 32 < K) {                     // register-prefetch next A frags
      a0 = load_a_frag(Arow0 + kb + 32, lane);
      a1 = load_a_frag(Arow1 + kb + 32, lane);
    }

    const bf16_t* bbase = &Bbuf[cur][0];
#pragma unroll
    for (int j = 0; j < 4; ++j) {
      v16bf b = load_b_frag(bbase + (j * 16 + nloc) * 32, lane);
      acc[0][j] = __builtin_amdgcn_wmma_f32_16x16x32_bf16(
          false, a0u, false, b, (short)0, acc[0][j], false, false);
      acc[1][j] = __builtin_amdgcn_wmma_f32_16x16x32_bf16(
          false, a1u, false, b, (short)0, acc[1][j], false, false);
    }

    __syncthreads();                       // everyone done reading buf[cur]
    if (kb + 64 < K)                       // refill buf[cur] for kb+64
      async_load_b128(cur ? ldsoff1 : ldsoff0, bsrc + kb + 64);
  }

#pragma unroll
  for (int m = 0; m < 2; ++m) {
#pragma unroll
    for (int j = 0; j < 4; ++j) {
#pragma unroll
      for (int r = 0; r < 8; ++r) {
        const int row = tm + m * 16 + r + 8 * hi;
        const int col = tn + j * 16 + nloc;
        float v = acc[m][j][r];
        if (doClip) v = fminf(fmaxf(v, -DB_CLIP), DB_CLIP);
        if (residual) v += residual[(size_t)row * N + col];
        C[(size_t)row * N + col] = v;
      }
    }
  }
}

// ---------------------------------------------------------------------------
// Kernel 4: RoPE + split into Q [B,Hq,S,hd] (scaled by 1/sqrt(hd)),
// K [B,Hkv,S,hd], V^T [B,Hkv,hd,S]   (all bf16)
// ---------------------------------------------------------------------------
__global__ void k_rope_split(const float* __restrict__ qkv, const int* __restrict__ pos,
                             const float* __restrict__ rsin, const float* __restrict__ rcos,
                             bf16_t* __restrict__ qd, bf16_t* __restrict__ kd,
                             bf16_t* __restrict__ vtd) {
  size_t idx = (size_t)blockIdx.x * blockDim.x + threadIdx.x;
  if (idx >= (size_t)DB_M * DB_QKVN) return;
  const int    c    = (int)(idx % DB_QKVN);
  const size_t rowi = idx / DB_QKVN;
  const int    s    = (int)(rowi & (DB_S - 1));
  const int    b    = (int)(rowi >> 11);
  const float* rowp = qkv + rowi * DB_QKVN;

  if (c < DB_D) {                               // Q
    const int h = c >> 7, d = c & 127;
    const int p = pos[rowi];
    const float x = rowp[c];
    const float partner = (d < 64) ? -rowp[c + 64] : rowp[c - 64];
    float val = x * rcos[p * DB_HD + d] + partner * rsin[p * DB_HD + d];
    val *= 0.08838834764831845f;                // 1/sqrt(128) folded into Q
    qd[(((size_t)b * DB_HQ + h) * DB_S + s) * DB_HD + d] = (bf16_t)val;
  } else if (c < DB_D + DB_HKV * DB_HD) {       // K
    const int cc = c - DB_D;
    const int h = cc >> 7, d = cc & 127;
    const int p = pos[rowi];
    const float x = rowp[c];
    const float partner = (d < 64) ? -rowp[c + 64] : rowp[c - 64];
    const float val = x * rcos[p * DB_HD + d] + partner * rsin[p * DB_HD + d];
    kd[(((size_t)b * DB_HKV + h) * DB_S + s) * DB_HD + d] = (bf16_t)val;
  } else {                                      // V (store transposed [hd][S])
    const int cc = c - (DB_D + DB_HKV * DB_HD);
    const int h = cc >> 7, d = cc & 127;
    vtd[(((size_t)b * DB_HKV + h) * DB_HD + d) * DB_S + s] = (bf16_t)rowp[c];
  }
}

// ---------------------------------------------------------------------------
// Kernel 5: causal GQA flash attention. Block = 8 waves sharing (b,h); wave w
// owns q-tile qgrp*8+w. K/V tiles (32 keys) staged in double-buffered LDS via
// global_load_async_to_lds_b128 and shared by all waves. Step count is made
// uniform across the block (that of the largest q-tile): fully-masked steps
// are numerically inert under online softmax (alpha=1, p=0).
// Output: attn [B,S, Hq*hd] bf16 (row-major [4096, 2048]).
// ---------------------------------------------------------------------------
__global__ void __launch_bounds__(256)
k_attention(const bf16_t* __restrict__ Q, const bf16_t* __restrict__ Kc,
            const bf16_t* __restrict__ VT, const int* __restrict__ amask,
            bf16_t* __restrict__ Obf) {
  __shared__ bf16_t Kbuf[2][32 * 128];   // [key][hd]   8 KB per buffer
  __shared__ bf16_t Vbuf[2][128 * 32];   // [hd][key]   8 KB per buffer
  __shared__ bf16_t Pbuf[8][16 * 32];    // per-wave P  8 KB

  const int tid  = threadIdx.x;
  const int wid  = tid >> 5;
  const int lane = tid & 31;
  const int qgrp = blockIdx.x & 15;           // 16 groups of 8 q-tiles
  const int h    = (blockIdx.x >> 4) & 15;
  const int b    = blockIdx.x >> 8;
  const int hkv  = h >> 2;                    // GROUPS = 4
  const int qt   = qgrp * 8 + wid;
  const int qbase = qt * 16;
  const int nloc  = lane & 15;
  const int hi    = lane >> 4;

  // Preload Q A-fragments for this wave's 16 rows (hd=128 -> 4 frags of K=32)
  const bf16_t* Qrow =
      Q + (((size_t)b * DB_HQ + h) * DB_S + (qbase + nloc)) * DB_HD;
  v16bf qfrag[4];
#pragma unroll
  for (int t = 0; t < 4; ++t) qfrag[t] = load_a_frag(Qrow + 32 * t, lane);

  const bf16_t* Kb  = Kc + ((size_t)b * DB_HKV + hkv) * DB_S * DB_HD;
  const bf16_t* VTb = VT + ((size_t)b * DB_HKV + hkv) * DB_HD * DB_S;
  const int* amr = amask + (size_t)b * DB_S;

  // Cooperative staging pieces (32 B per thread per tile, 2x b128 each):
  const int kkey = tid >> 3;                  // K tile: key row 0..31
  const int khd  = (tid & 7) * 16;            //         hd offset
  const bf16_t* ksrc = Kb + (size_t)kkey * DB_HD + khd;
  const unsigned klds[2] = {
      (unsigned)(uintptr_t)&Kbuf[0][kkey * 128 + khd],
      (unsigned)(uintptr_t)&Kbuf[1][kkey * 128 + khd] };
  const int vrow = tid >> 1;                  // V tile: hd row 0..127
  const int vko  = (tid & 1) * 16;            //         key offset
  const bf16_t* vsrc = VTb + (size_t)vrow * DB_S + vko;
  const unsigned vlds[2] = {
      (unsigned)(uintptr_t)&Vbuf[0][vrow * 32 + vko],
      (unsigned)(uintptr_t)&Vbuf[1][vrow * 32 + vko] };

#define ISSUE_TILE(buf, jb)                                        \
  do {                                                             \
    async_load_b128(klds[buf],      ksrc + (size_t)(jb) * DB_HD);  \
    async_load_b128(klds[buf] + 16, ksrc + (size_t)(jb) * DB_HD + 8); \
    async_load_b128(vlds[buf],      vsrc + (jb));                  \
    async_load_b128(vlds[buf] + 16, vsrc + (jb) + 8);              \
  } while (0)

  float mrow[8], lrow[8];
  v8f o[8];
#pragma unroll
  for (int r = 0; r < 8; ++r) { mrow[r] = DB_NEGINF; lrow[r] = 0.0f; }
#pragma unroll
  for (int n = 0; n < 8; ++n) o[n] = vzero8();

  // Uniform step count per block = that of the largest q-tile in the block.
  const int nsteps = ((qgrp * 8 + 7) >> 1) + 1;

  ISSUE_TILE(0, 0);
  if (nsteps > 1) ISSUE_TILE(1, 32);

  for (int stp = 0; stp < nsteps; ++stp) {
    const int jb  = stp * 32;
    const int cur = stp & 1;
    if (stp + 1 < nsteps) asm volatile("s_wait_asynccnt 0x4" ::: "memory");
    else                  asm volatile("s_wait_asynccnt 0x0" ::: "memory");
    __syncthreads();                          // K/V tile visible block-wide

    // ---- scores: two 16x16 tiles, K=hd=128 accumulated over 4 WMMAs each
    v8f s0 = vzero8(), s1 = vzero8();
    const bf16_t* K0 = &Kbuf[cur][(size_t)nloc * 128];
    const bf16_t* K1 = &Kbuf[cur][(size_t)(16 + nloc) * 128];
#pragma unroll
    for (int ks = 0; ks < 4; ++ks) {
      v16bf b0 = load_b_frag(K0 + ks * 32, lane);
      s0 = __builtin_amdgcn_wmma_f32_16x16x32_bf16(false, qfrag[ks], false, b0,
                                                   (short)0, s0, false, false);
    }
#pragma unroll
    for (int ks = 0; ks < 4; ++ks) {
      v16bf b1 = load_b_frag(K1 + ks * 32, lane);
      s1 = __builtin_amdgcn_wmma_f32_16x16x32_bf16(false, qfrag[ks], false, b1,
                                                   (short)0, s1, false, false);
    }

    const int key0 = jb + nloc;
    const int key1 = jb + 16 + nloc;
    const bool am0 = amr[key0] != 0;
    const bool am1 = amr[key1] != 0;

    // ---- online softmax (rows r+8*hi live across each 16-lane half)
#pragma unroll
    for (int r = 0; r < 8; ++r) {
      const int row = qbase + r + 8 * hi;
      float a0 = (am0 && key0 <= row) ? s0[r] : DB_NEGINF;
      float a1 = (am1 && key1 <= row) ? s1[r] : DB_NEGINF;
      float mx = hred_max16(fmaxf(a0, a1));
      const float mn = fmaxf(mrow[r], mx);
      const float alpha = __expf(mrow[r] - mn);
      const float p0 = __expf(a0 - mn);
      const float p1 = __expf(a1 - mn);
      lrow[r] = lrow[r] * alpha + hred_sum16(p0 + p1);
      mrow[r] = mn;
#pragma unroll
      for (int n = 0; n < 8; ++n) o[n][r] *= alpha;
      Pbuf[wid][(r + 8 * hi) * 32 + nloc]      = (bf16_t)p0;
      Pbuf[wid][(r + 8 * hi) * 32 + 16 + nloc] = (bf16_t)p1;
    }

    // ---- O += P(16x32) @ V(32x128): P in A-frag layout from per-wave LDS
    v16bf pa = load_a_frag(&Pbuf[wid][nloc * 32], lane);
#pragma unroll
    for (int n = 0; n < 8; ++n) {
      v16bf vb = load_b_frag(&Vbuf[cur][(size_t)(n * 16 + nloc) * 32], lane);
      o[n] = __builtin_amdgcn_wmma_f32_16x16x32_bf16(false, pa, false, vb,
                                                     (short)0, o[n], false, false);
    }

    __syncthreads();                          // block done reading buf[cur]
    if (stp + 2 < nsteps) ISSUE_TILE(cur, jb + 64);
  }
#undef ISSUE_TILE

  // ---- epilogue: normalize by l, write bf16 [4096, 2048]
  float linv[8];
#pragma unroll
  for (int r = 0; r < 8; ++r) linv[r] = 1.0f / lrow[r];
#pragma unroll
  for (int n = 0; n < 8; ++n) {
#pragma unroll
    for (int r = 0; r < 8; ++r) {
      const int row = qbase + r + 8 * hi;
      const int col = h * DB_HD + n * 16 + nloc;
      Obf[((size_t)b * DB_S + row) * DB_D + col] = (bf16_t)(o[n][r] * linv[r]);
    }
  }
}

// ---------------------------------------------------------------------------
// Launcher
// ---------------------------------------------------------------------------
extern "C" void kernel_launch(void* const* d_in, const int* in_sizes, int n_in,
                              void* d_out, int out_size, void* d_ws, size_t ws_size,
                              hipStream_t stream) {
  const float* hidden = (const float*)d_in[0];
  const int*   amask  = (const int*)d_in[1];
  const int*   pos    = (const int*)d_in[2];
  const float* wqkv   = (const float*)d_in[3];
  const float* wout   = (const float*)d_in[4];
  const float* n1w    = (const float*)d_in[5];
  const float* n2w    = (const float*)d_in[6];
  const float* rsin   = (const float*)d_in[7];
  const float* rcos   = (const float*)d_in[8];

  float* out_resid = (float*)d_out;                          // [4096,2048]
  float* out_post  = (float*)d_out + (size_t)DB_M * DB_D;    // [4096,2048]

  char* ws = (char*)d_ws;
  bf16_t* xnorm  = (bf16_t*)ws;  ws += (size_t)DB_M * DB_D * 2;          // 16 MB
  bf16_t* wqkvT  = (bf16_t*)ws;  ws += (size_t)DB_QKVN * DB_D * 2;       // 12 MB
  bf16_t* woutT  = (bf16_t*)ws;  ws += (size_t)DB_D * DB_D * 2;          //  8 MB
  float*  qkvf   = (float*)ws;   ws += (size_t)DB_M * DB_QKVN * 4;       // 48 MB
  bf16_t* qbf    = (bf16_t*)ws;  ws += (size_t)DB_M * DB_D * 2;          // 16 MB
  bf16_t* kbf    = (bf16_t*)ws;  ws += (size_t)DB_B * DB_HKV * DB_S * DB_HD * 2;
  bf16_t* vtbf   = (bf16_t*)ws;  ws += (size_t)DB_B * DB_HKV * DB_S * DB_HD * 2;
  bf16_t* attnbf = (bf16_t*)ws;  ws += (size_t)DB_M * DB_D * 2;          // 16 MB

  // 1) weight convert+transpose to bf16 [N][K]
  {
    size_t n1 = (size_t)DB_D * DB_QKVN;
    k_convtr<<<(int)((n1 + 255) / 256), 256, 0, stream>>>(wqkv, wqkvT, DB_D, DB_QKVN);
    size_t n2 = (size_t)DB_D * DB_D;
    k_convtr<<<(int)((n2 + 255) / 256), 256, 0, stream>>>(wout, woutT, DB_D, DB_D);
  }
  // 2) pre-norm -> bf16
  k_rmsnorm_bf16<<<DB_M, 256, 0, stream>>>(hidden, n1w, xnorm);
  // 3) QKV GEMM (M=4096, N=3072, K=2048) + clip
  {
    const int nblk64 = DB_QKVN / 64;                         // 48
    const int blocks = (DB_M / 256) * nblk64;                // 768
    k_gemm_bf16<<<blocks, 256, 0, stream>>>(xnorm, wqkvT, qkvf, nullptr,
                                            DB_M, DB_QKVN, DB_D, nblk64, 1);
  }
  // 4) RoPE + split (Q scaled by 1/sqrt(hd), V stored transposed)
  {
    size_t n = (size_t)DB_M * DB_QKVN;
    k_rope_split<<<(int)((n + 255) / 256), 256, 0, stream>>>(qkvf, pos, rsin, rcos,
                                                             qbf, kbf, vtbf);
  }
  // 5) causal GQA flash attention (block = 8 q-tiles of one (b,h))
  {
    const int blocks = DB_B * DB_HQ * 16;                    // 512
    k_attention<<<blocks, 256, 0, stream>>>(qbf, kbf, vtbf, amask, attnbf);
  }
  // 6) out-proj GEMM + residual add -> residual_out
  {
    const int nblk64 = DB_D / 64;                            // 32
    const int blocks = (DB_M / 256) * nblk64;                // 512
    k_gemm_bf16<<<blocks, 256, 0, stream>>>(attnbf, woutT, out_resid, hidden,
                                            DB_M, DB_D, DB_D, nblk64, 0);
  }
  // 7) post-norm -> out_post
  k_rmsnorm_f32<<<DB_M, 256, 0, stream>>>(out_resid, n2w, out_post);
}
